// RTModule_82471962017877
// MI455X (gfx1250) — matrix operand order repaired and checked by reading, not compiled, not run on hardware
//
#include <hip/hip_runtime.h>
#include <hip/hip_bf16.h>

#define B_  2
#define C_  256
#define N_  2304      // 48*48
#define NH_ 4
#define DK_ 64
#define C2_ 512

typedef __attribute__((ext_vector_type(16))) _Float16 v16h;
typedef __attribute__((ext_vector_type(8)))  float    f32x8;

union frag_u {
    uint4 q[2];
    unsigned short s[16];
    v16h v;
};

__device__ __forceinline__ unsigned short f2h(float f) {
    _Float16 h = (_Float16)f;
    return __builtin_bit_cast(unsigned short, h);
}

// A-fragment (16x32 f16, MxK): row pointer `base` (16B-aligned chunks), lane-half offset s in {0,8}.
// VGPR0..3 hold K = s+0..7 (packed pairs), VGPR4..7 hold K = s+16..23.
__device__ __forceinline__ v16h load_fragA(const unsigned short* base, int s) {
    frag_u f;
    f.q[0] = *(const uint4*)(base + s);
    f.q[1] = *(const uint4*)(base + s + 16);
    return f.v;
}

// B-fragment (32x16 f16, KxN) from a K-contiguous (transposed) row: kb in {0,16}.
// Lanes 0-15: K = 0..7,16..23 of column n=lane; lanes 16-31: K = 8..15,24..31.
__device__ __forceinline__ v16h load_fragB(const unsigned short* base, int kb) {
    frag_u f;
    f.q[0] = *(const uint4*)(base + kb / 2);
    f.q[1] = *(const uint4*)(base + kb / 2 + 16);
    return f.v;
}

__device__ __forceinline__ f32x8 wmma_f16(v16h a, v16h b, f32x8 c) {
    return __builtin_amdgcn_wmma_f32_16x16x32_f16(false, a, false, b, (short)0, c, false, false);
}

// ---------------------------------------------------------------- pooling
__global__ __launch_bounds__(256) void pool_kernel(const float* __restrict__ x,
                                                   const float* __restrict__ y,
                                                   float* __restrict__ pooled) {
    int bc = blockIdx.x;            // B*512 blocks
    int b  = bc >> 9;
    int c2 = bc & 511;
    const float* src = (c2 < C_) ? (x + ((size_t)b * C_ + c2) * N_)
                                 : (y + ((size_t)b * C_ + (c2 - C_)) * N_);
    float s = 0.f;
    for (int i = threadIdx.x; i < N_; i += 256) s += src[i];
    __shared__ float red[256];
    red[threadIdx.x] = s;
    __syncthreads();
    for (int o = 128; o > 0; o >>= 1) {
        if (threadIdx.x < o) red[threadIdx.x] += red[threadIdx.x + o];
        __syncthreads();
    }
    if (threadIdx.x == 0) pooled[bc] = red[0] * (1.0f / N_);
}

// ---------------------------------------------------------------- SE MLP
__global__ __launch_bounds__(128) void se_kernel(const float* __restrict__ pooled,
                                                 const float* __restrict__ f1w,
                                                 const float* __restrict__ f1b,
                                                 const float* __restrict__ f2w,
                                                 const float* __restrict__ f2b,
                                                 float* __restrict__ g) {
    int b = blockIdx.x;
    __shared__ float hid[64];
    const float* p = pooled + b * C2_;
    if (threadIdx.x < 64) {
        float a = f1b[threadIdx.x];
        const float* wr = f1w + threadIdx.x * C2_;
        for (int c = 0; c < C2_; c++) a += wr[c] * p[c];
        hid[threadIdx.x] = fmaxf(a, 0.f);
    }
    __syncthreads();
    for (int c2 = threadIdx.x; c2 < C2_; c2 += 128) {
        float a = f2b[c2];
        const float* wr = f2w + c2 * 64;
        #pragma unroll 8
        for (int j = 0; j < 64; j++) a += wr[j] * hid[j];
        g[b * C2_ + c2] = 1.f / (1.f + __expf(-a));
    }
}

// ---------------------------------------------------------------- SE apply
__global__ __launch_bounds__(256) void scale_kernel(const float* __restrict__ x,
                                                    const float* __restrict__ y,
                                                    const float* __restrict__ g,
                                                    float* __restrict__ xs,
                                                    float* __restrict__ ys) {
    int i = blockIdx.x * 256 + threadIdx.x;   // over B*C*N
    int b = i / (C_ * N_);
    int c = (i / N_) % C_;
    float gx = g[b * C2_ + c];
    float gy = g[b * C2_ + C_ + c];
    xs[i] = x[i] * (1.f + gx);
    ys[i] = y[i] * (1.f + gy);
}

// ---------------------------------------------------------------- conv1x1 GEMM: O[o,n] = sum_c W[o,c]*X[c,n] + bias[o]
__global__ __launch_bounds__(256) void gemm_conv1x1(const float* __restrict__ Wm,
                                                    const float* __restrict__ bias,
                                                    const float* __restrict__ X,
                                                    float* __restrict__ O) {
    const int NB = N_ / 128;                 // 18
    const int nb = blockIdx.x % NB;
    const int b  = blockIdx.x / NB;
    const int mB = blockIdx.y * 64;
    const int n0 = nb * 128;
    const float* Xb = X + (size_t)b * C_ * N_;
    float* Ob = O + (size_t)b * C_ * N_;

    __shared__ unsigned short Wt[64][40];    // [m][k], row stride 80B
    __shared__ unsigned short Xt[128][40];   // [n][k] (transposed), row stride 80B

    const int tid  = threadIdx.x;
    const int lane = tid & 31;
    const int wave = tid >> 5;               // 0..7
    const int wm   = (wave >> 2) * 32;       // 0,32
    const int wn   = (wave & 3) * 32;        // 0..96
    const int sA   = ((lane >> 4) & 1) * 8;
    const int kbB  = (lane < 16) ? 0 : 16;

    f32x8 acc[2][2] = {};

    for (int kk = 0; kk < C_; kk += 32) {
        // stage W tile 64x32
        {
            int m  = tid >> 2;
            int k0 = (tid & 3) * 8;
            const float* src = Wm + (mB + m) * C_ + kk + k0;
            float4 a = *(const float4*)(src);
            float4 c = *(const float4*)(src + 4);
            unsigned short* dst = &Wt[m][k0];
            dst[0] = f2h(a.x); dst[1] = f2h(a.y); dst[2] = f2h(a.z); dst[3] = f2h(a.w);
            dst[4] = f2h(c.x); dst[5] = f2h(c.y); dst[6] = f2h(c.z); dst[7] = f2h(c.w);
        }
        // stage X tile 32x128 transposed into Xt[n][k]
        {
            int nl = (tid & 31) * 4;
            int kl = (tid >> 5) * 4;
            const float* s0 = Xb + (size_t)(kk + kl + 0) * N_ + n0 + nl;
            const float* s1 = Xb + (size_t)(kk + kl + 1) * N_ + n0 + nl;
            const float* s2 = Xb + (size_t)(kk + kl + 2) * N_ + n0 + nl;
            const float* s3 = Xb + (size_t)(kk + kl + 3) * N_ + n0 + nl;
            float4 r0 = *(const float4*)s0;
            float4 r1 = *(const float4*)s1;
            float4 r2 = *(const float4*)s2;
            float4 r3 = *(const float4*)s3;
            unsigned short* d0 = &Xt[nl + 0][kl];
            d0[0] = f2h(r0.x); d0[1] = f2h(r1.x); d0[2] = f2h(r2.x); d0[3] = f2h(r3.x);
            unsigned short* d1 = &Xt[nl + 1][kl];
            d1[0] = f2h(r0.y); d1[1] = f2h(r1.y); d1[2] = f2h(r2.y); d1[3] = f2h(r3.y);
            unsigned short* d2 = &Xt[nl + 2][kl];
            d2[0] = f2h(r0.z); d2[1] = f2h(r1.z); d2[2] = f2h(r2.z); d2[3] = f2h(r3.z);
            unsigned short* d3 = &Xt[nl + 3][kl];
            d3[0] = f2h(r0.w); d3[1] = f2h(r1.w); d3[2] = f2h(r2.w); d3[3] = f2h(r3.w);
            if (kk + 32 < C_)
                __builtin_prefetch(Xb + (size_t)(kk + 32 + kl) * N_ + n0 + nl, 0, 1);
        }
        __syncthreads();

        v16h aF0 = load_fragA(&Wt[wm +      (lane & 15)][0], sA);
        v16h aF1 = load_fragA(&Wt[wm + 16 + (lane & 15)][0], sA);
        v16h bF0 = load_fragB(&Xt[wn +      (lane & 15)][0], kbB);
        v16h bF1 = load_fragB(&Xt[wn + 16 + (lane & 15)][0], kbB);

        acc[0][0] = wmma_f16(aF0, bF0, acc[0][0]);
        acc[0][1] = wmma_f16(aF0, bF1, acc[0][1]);
        acc[1][0] = wmma_f16(aF1, bF0, acc[1][0]);
        acc[1][1] = wmma_f16(aF1, bF1, acc[1][1]);
        __syncthreads();
    }

    const int half8 = (lane < 16) ? 0 : 8;
    #pragma unroll
    for (int i = 0; i < 2; i++)
        #pragma unroll
        for (int j = 0; j < 2; j++) {
            int ncol = n0 + wn + j * 16 + (lane & 15);
            #pragma unroll
            for (int v = 0; v < 8; v++) {
                int m = mB + wm + i * 16 + v + half8;
                Ob[(size_t)m * N_ + ncol] = acc[i][j][v] + bias[m];
            }
        }
}

// ---------------------------------------------------------------- softmax row stats: max_n and sumexp_n of S = (Q^T K)/8
__global__ __launch_bounds__(128) void rowstats_kernel(const float* __restrict__ Q,
                                                       const float* __restrict__ K,
                                                       float* __restrict__ smax,
                                                       float* __restrict__ ssum) {
    int idx = blockIdx.x;                    // B*NH*(N/64) = 288
    int ng  = idx % (N_ / 64);  idx /= (N_ / 64);
    int h   = idx % NH_;
    int b   = idx / NH_;
    const float* Qh = Q + ((size_t)b * C_ + h * DK_) * N_;
    const float* Kh = K + ((size_t)b * C_ + h * DK_) * N_;

    int tid = threadIdx.x, lane = tid & 31, wave = tid >> 5;
    int nrow0 = ng * 64 + wave * 16;
    const int sA  = ((lane >> 4) & 1) * 8;
    const int kbB = (lane < 16) ? 0 : 16;

    // A fragments: Q^T rows n, K-dim = dk (two 32 chunks). Loaded once.
    v16h aF[2];
    {
        int n = nrow0 + (lane & 15);
        #pragma unroll
        for (int c = 0; c < 2; c++) {
            frag_u f;
            #pragma unroll
            for (int r = 0; r < 8; r++) f.s[r]     = f2h(Qh[(size_t)(c * 32 + sA + r) * N_ + n]);
            #pragma unroll
            for (int r = 0; r < 8; r++) f.s[8 + r] = f2h(Qh[(size_t)(c * 32 + sA + 16 + r) * N_ + n]);
            aF[c] = f.v;
        }
    }

    __shared__ unsigned short Kt[16][72];    // [m][dd], row stride 144B

    float runmax[8], runsum[8];
    #pragma unroll
    for (int v = 0; v < 8; v++) { runmax[v] = -3.4e38f; runsum[v] = 0.f; }

    for (int mt = 0; mt < N_; mt += 16) {
        {   // stage K tile 16(m) x 64(dd), transposed (dd contiguous)
            int m  = tid >> 3;
            int d0 = (tid & 7) * 8;
            unsigned short* dst = &Kt[m][d0];
            #pragma unroll
            for (int r = 0; r < 8; r++) dst[r] = f2h(Kh[(size_t)(d0 + r) * N_ + mt + m]);
        }
        __syncthreads();
        f32x8 sacc = {};
        #pragma unroll
        for (int c = 0; c < 2; c++) {
            v16h bF = load_fragB(&Kt[lane & 15][c * 32], kbB);
            sacc = wmma_f16(aF[c], bF, sacc);
        }
        #pragma unroll
        for (int v = 0; v < 8; v++) {
            float s  = sacc[v] * 0.125f;
            float nm = fmaxf(runmax[v], s);
            runsum[v] = runsum[v] * __expf(runmax[v] - nm) + __expf(s - nm);
            runmax[v] = nm;
        }
        __syncthreads();
    }
    // cross-lane reduction within each 16-lane half (each half owns rows v / v+8)
    #pragma unroll
    for (int v = 0; v < 8; v++) {
        float m = runmax[v], sm = runsum[v];
        for (int off = 1; off < 16; off <<= 1) {
            float om = __shfl_xor(m, off, 16);
            float os = __shfl_xor(sm, off, 16);
            float nm = fmaxf(m, om);
            sm = sm * __expf(m - nm) + os * __expf(om - nm);
            m = nm;
        }
        runmax[v] = m; runsum[v] = sm;
    }
    if ((lane & 15) == 0) {
        int half = lane >> 4;
        #pragma unroll
        for (int v = 0; v < 8; v++) {
            int n = nrow0 + v + half * 8;
            int o = (b * NH_ + h) * N_ + n;
            smax[o] = runmax[v];
            ssum[o] = runsum[v];
        }
    }
}

// ---------------------------------------------------------------- attention output: O[dd,m] = sum_n V[dd,n] * P[n,m]
__global__ __launch_bounds__(256) void attnout_kernel(const float* __restrict__ Q,
                                                      const float* __restrict__ K,
                                                      const float* __restrict__ V,
                                                      const float* __restrict__ smax,
                                                      const float* __restrict__ ssum,
                                                      float* __restrict__ O) {
    int idx = blockIdx.x;                    // B*NH*18 = 144
    int mg  = idx % (N_ / 128);  idx /= (N_ / 128);
    int h   = idx % NH_;
    int b   = idx / NH_;
    const float* Qh = Q + ((size_t)b * C_ + h * DK_) * N_;
    const float* Kh = K + ((size_t)b * C_ + h * DK_) * N_;
    const float* Vh = V + ((size_t)b * C_ + h * DK_) * N_;
    const float* mx = smax + (b * NH_ + h) * N_;
    const float* sm = ssum + (b * NH_ + h) * N_;
    float* Oh = O + ((size_t)b * C_ + h * DK_) * N_;

    int tid = threadIdx.x, lane = tid & 31, wave = tid >> 5;
    int m0 = mg * 128 + wave * 16;           // this wave's 16 output columns
    const int sA   = ((lane >> 4) & 1) * 8;
    const int kbB  = (lane < 16) ? 0 : 16;
    const int half8 = (lane < 16) ? 0 : 8;
    const int mloc = lane & 15;

    __shared__ unsigned short Qt[32][72];    // [n][dd]
    __shared__ unsigned short Vt[64][40];    // [dd][n]
    __shared__ unsigned short Pt[8][16][32]; // per-wave [m][n]
    __shared__ float Sst[2][32];             // rowmax, 1/rowsum for current n chunk

    // K B-fragments for scores (fixed m columns): load once, strided global reads.
    v16h kF[2];
    {
        int m = m0 + mloc;
        #pragma unroll
        for (int c = 0; c < 2; c++) {
            frag_u f;
            #pragma unroll
            for (int r = 0; r < 16; r++) f.s[r] = f2h(Kh[(size_t)(c * 32 + kbB / 2 + r + (r & 8)) * N_ + m]);
            kF[c] = f.v;
        }
    }

    f32x8 oacc[4] = {};

    for (int n0 = 0; n0 < N_; n0 += 32) {
        {   // stage Q^T chunk 32(n) x 64(dd)
            int nl = tid & 31, d0 = (tid >> 5) * 8;
            unsigned short* dst = &Qt[nl][d0];
            #pragma unroll
            for (int r = 0; r < 8; r++) dst[r] = f2h(Qh[(size_t)(d0 + r) * N_ + n0 + nl]);
        }
        {   // stage V chunk 64(dd) x 32(n)
            int dd = tid >> 2, nl0 = (tid & 3) * 8;
            const float* src = Vh + (size_t)dd * N_ + n0 + nl0;
            float4 a = *(const float4*)(src);
            float4 c = *(const float4*)(src + 4);
            unsigned short* dst = &Vt[dd][nl0];
            dst[0] = f2h(a.x); dst[1] = f2h(a.y); dst[2] = f2h(a.z); dst[3] = f2h(a.w);
            dst[4] = f2h(c.x); dst[5] = f2h(c.y); dst[6] = f2h(c.z); dst[7] = f2h(c.w);
        }
        if (tid < 32) {
            Sst[0][tid] = mx[n0 + tid];
            Sst[1][tid] = 1.f / sm[n0 + tid];
        }
        __syncthreads();

        // scores for the two 16-row n sub-tiles, converted to P, written transposed
        #pragma unroll
        for (int sub = 0; sub < 2; sub++) {
            f32x8 sacc = {};
            #pragma unroll
            for (int c = 0; c < 2; c++) {
                v16h aF = load_fragA(&Qt[sub * 16 + mloc][c * 32], sA);
                sacc = wmma_f16(aF, kF[c], sacc);
            }
            #pragma unroll
            for (int v = 0; v < 8; v += 2) {
                int na = sub * 16 + v + half8;     // even
                float p0 = __expf(sacc[v]     * 0.125f - Sst[0][na])     * Sst[1][na];
                float p1 = __expf(sacc[v + 1] * 0.125f - Sst[0][na + 1]) * Sst[1][na + 1];
                unsigned int pk = (unsigned int)f2h(p0) | ((unsigned int)f2h(p1) << 16);
                *(unsigned int*)&Pt[wave][mloc][na] = pk;
            }
        }
        __syncthreads();

        // O tile accumulation: A = V rows dd, B = P rows n
        {
            v16h bF = load_fragB(&Pt[wave][mloc][0], kbB);
            #pragma unroll
            for (int d = 0; d < 4; d++) {
                v16h aF = load_fragA(&Vt[d * 16 + mloc][0], sA);
                oacc[d] = wmma_f16(aF, bF, oacc[d]);
            }
        }
        __syncthreads();
    }

    #pragma unroll
    for (int d = 0; d < 4; d++)
        #pragma unroll
        for (int v = 0; v < 8; v++) {
            int dd = d * 16 + v + half8;
            Oh[(size_t)dd * N_ + m0 + mloc] = oacc[d][v];
        }
}

// ---------------------------------------------------------------- final gating + residual
__global__ __launch_bounds__(256) void gate_kernel(const float* __restrict__ xn,
                                                   const float* __restrict__ yn,
                                                   const float* __restrict__ gw,
                                                   const float* __restrict__ gb,
                                                   float* __restrict__ out) {
    int i = blockIdx.x * 256 + threadIdx.x;  // over B*N pixels
    int b = i / N_;
    int pix = i % N_;
    const float* xb = xn + (size_t)b * C_ * N_ + pix;
    const float* yb = yn + (size_t)b * C_ * N_ + pix;
    float a0 = gb[0], a1 = gb[1];
    for (int c = 0; c < C_; c++) {
        float xv = xb[(size_t)c * N_], yv = yb[(size_t)c * N_];
        a0 += gw[c] * xv + gw[C_ + c] * yv;
        a1 += gw[C2_ + c] * xv + gw[C2_ + C_ + c] * yv;
    }
    float g0 = 1.f / (1.f + __expf(-a0));
    float g1 = 1.f / (1.f + __expf(-a1));
    float* ox = out + (size_t)b * C_ * N_ + pix;
    float* oy = out + (size_t)B_ * C_ * N_ + (size_t)b * C_ * N_ + pix;
    for (int c = 0; c < C_; c++) {
        ox[(size_t)c * N_] = xb[(size_t)c * N_] * (1.f + g0);
        oy[(size_t)c * N_] = yb[(size_t)c * N_] * (1.f + g1);
    }
}

extern "C" void kernel_launch(void* const* d_in, const int* in_sizes, int n_in,
                              void* d_out, int out_size, void* d_ws, size_t ws_size,
                              hipStream_t stream) {
    const float* x      = (const float*)d_in[0];
    const float* y      = (const float*)d_in[1];
    const float* f1w    = (const float*)d_in[2];
    const float* f1b    = (const float*)d_in[3];
    const float* f2w    = (const float*)d_in[4];
    const float* f2b    = (const float*)d_in[5];
    const float* t1_qw  = (const float*)d_in[6];
    const float* t1_qb  = (const float*)d_in[7];
    const float* t1_kw  = (const float*)d_in[8];
    const float* t1_kb  = (const float*)d_in[9];
    const float* t1_vw  = (const float*)d_in[10];
    const float* t1_vb  = (const float*)d_in[11];
    const float* t2_qw  = (const float*)d_in[12];
    const float* t2_qb  = (const float*)d_in[13];
    const float* t2_kw  = (const float*)d_in[14];
    const float* t2_kb  = (const float*)d_in[15];
    const float* t2_vw  = (const float*)d_in[16];
    const float* t2_vb  = (const float*)d_in[17];
    const float* gate_w = (const float*)d_in[18];
    const float* gate_b = (const float*)d_in[19];
    float* out = (float*)d_out;

    const size_t BCN = (size_t)B_ * C_ * N_;
    float* ws     = (float*)d_ws;
    float* pooled = ws;                 // B*512
    float* gsig   = ws + 1024;          // B*512
    float* xs     = ws + 2048;          // BCN
    float* ysb    = xs + BCN;
    float* qb     = ysb + BCN;
    float* kb     = qb + BCN;
    float* vb     = kb + BCN;
    float* xnw    = vb + BCN;
    float* ynw    = xnw + BCN;
    float* smaxw  = ynw + BCN;          // B*NH*N
    float* ssumw  = smaxw + (size_t)B_ * NH_ * N_;

    const dim3 gemmGrid((N_ / 128) * B_, C_ / 64);

    // SE scaling
    pool_kernel<<<B_ * C2_, 256, 0, stream>>>(x, y, pooled);
    se_kernel<<<B_, 128, 0, stream>>>(pooled, f1w, f1b, f2w, f2b, gsig);
    scale_kernel<<<(int)(BCN / 256), 256, 0, stream>>>(x, y, gsig, xs, ysb);

    // RTrans 1: q=xs, k=v=ys
    gemm_conv1x1<<<gemmGrid, 256, 0, stream>>>(t1_qw, t1_qb, xs, qb);
    gemm_conv1x1<<<gemmGrid, 256, 0, stream>>>(t1_kw, t1_kb, ysb, kb);
    gemm_conv1x1<<<gemmGrid, 256, 0, stream>>>(t1_vw, t1_vb, ysb, vb);
    rowstats_kernel<<<B_ * NH_ * (N_ / 64), 128, 0, stream>>>(qb, kb, smaxw, ssumw);
    attnout_kernel<<<B_ * NH_ * (N_ / 128), 256, 0, stream>>>(qb, kb, vb, smaxw, ssumw, xnw);

    // RTrans 2: q=ys, k=v=x_new
    gemm_conv1x1<<<gemmGrid, 256, 0, stream>>>(t2_qw, t2_qb, ysb, qb);
    gemm_conv1x1<<<gemmGrid, 256, 0, stream>>>(t2_kw, t2_kb, xnw, kb);
    gemm_conv1x1<<<gemmGrid, 256, 0, stream>>>(t2_vw, t2_vb, xnw, vb);
    rowstats_kernel<<<B_ * NH_ * (N_ / 64), 128, 0, stream>>>(qb, kb, smaxw, ssumw);
    attnout_kernel<<<B_ * NH_ * (N_ / 128), 256, 0, stream>>>(qb, kb, vb, smaxw, ssumw, ynw);

    // gating + residual into d_out
    gate_kernel<<<(B_ * N_) / 256, 256, 0, stream>>>(xnw, ynw, gate_w, gate_b, out);
}
